// BertSelfAttention_17609365913815
// MI455X (gfx1250) — matrix operand order repaired and checked by reading, not compiled, not run on hardware
//
#include <hip/hip_runtime.h>
#include <hip/hip_bf16.h>
#include <cstdint>

// ---------------------------------------------------------------------------
// BERT self-attention for MI455X (gfx1250), wave32, WMMA f32<=f16 path.
//   B=2, S=2048, H=1024, NH=16, HD=64
// Pipeline:
//   k_cvt      : fp32 -> fp16 copies of X, Wq, Wk, Wv (all fit in 192MB L2)
//   k_qkv_proj : WMMA GEMM  Q/K -> [b,h,s,64] fp16,  V -> transposed [b,h,64,s]
//   k_flash    : fused flash-attention; K/V blocks double-buffered into LDS
//                via GLOBAL_LOAD_ASYNC_TO_LDS_B128 (ASYNCcnt pipeline),
//                scores never hit memory.
// ---------------------------------------------------------------------------

typedef _Float16 v8h  __attribute__((ext_vector_type(8)));
typedef _Float16 v16h __attribute__((ext_vector_type(16)));
typedef float    v8f  __attribute__((ext_vector_type(8)));

#define BB 2
#define SS 2048
#define HH 1024
#define NH 16
#define HD 64

static __device__ __forceinline__ v16h pack16(v8h lo, v8h hi) {
  v16h r;
#pragma unroll
  for (int i = 0; i < 8; ++i) { r[i] = lo[i]; r[8 + i] = hi[i]; }
  return r;
}

static __device__ __forceinline__ v8f wmma_f16(v16h a, v16h b, v8f c) {
  // D = A(16x32,f16) * B(32x16,f16) + C(16x16,f32)
  return __builtin_amdgcn_wmma_f32_16x16x32_f16(
      /*neg_a=*/false, a, /*neg_b=*/false, b,
      /*c_mod=*/(short)0, c, /*reuse_a=*/false, /*reuse_b=*/false);
}

// Async copy 16B from global to LDS (per lane). Tracked by ASYNCcnt.
static __device__ __forceinline__ void async_ld_b128(uint32_t lds_addr,
                                                     const void* gaddr) {
  asm volatile("global_load_async_to_lds_b128 %0, %1, off"
               :: "v"(lds_addr), "v"(gaddr) : "memory");
}
static __device__ __forceinline__ void wait_async0() {
  asm volatile("s_wait_asynccnt 0x0" ::: "memory");
}

// ---------------------------------------------------------------------------
// fp32 -> fp16 conversion (grid-stride)
// ---------------------------------------------------------------------------
__global__ __launch_bounds__(256) void k_cvt(const float* __restrict__ src,
                                             _Float16* __restrict__ dst, int n) {
  int i = blockIdx.x * blockDim.x + threadIdx.x;
  int stride = gridDim.x * blockDim.x;
  for (; i < n; i += stride) dst[i] = (_Float16)src[i];
}

// ---------------------------------------------------------------------------
// QKV projection:  out[m,n] = sum_k X[m,k] * W[n,k] + bias[n]
//   M = B*S = 4096, N = H = 1024, K = H = 1024
// Block: 256 threads = 8 waves. Wave tile: 16(M) x 64(N), K-loop step 32.
// grid = (M/128, N/64, 3)   z selects Q / K / V.
// Q,K written [b,h,s,64]; V written transposed [b,h,64,s].
// ---------------------------------------------------------------------------
__global__ __launch_bounds__(256) void k_qkv_proj(
    const _Float16* __restrict__ X,   // [4096][1024]
    const _Float16* __restrict__ W,   // [3][1024][1024]
    const float* __restrict__ bq, const float* __restrict__ bk,
    const float* __restrict__ bv,
    _Float16* __restrict__ Q, _Float16* __restrict__ K,
    _Float16* __restrict__ Vt) {
  const int lane  = threadIdx.x & 31;
  const int wave  = threadIdx.x >> 5;
  const int which = blockIdx.z;
  const int m0 = (blockIdx.x * 8 + wave) * 16;
  const int n0 = blockIdx.y * 64;

  const _Float16* Wp   = W + (size_t)which * HH * HH;
  const float*    bias = (which == 0) ? bq : ((which == 1) ? bk : bv);

  const int ncol  = lane & 15;
  const int half  = lane >> 4;
  const int hoffA = half * 8;    // A-frag K sub-offset (16-bit A 16x32 layout)
  const int koffB = half * 16;   // unused; the B fragment uses hoffA (same K order as A)

  v8f acc[4] = {};

  const _Float16* xrow = X + (size_t)(m0 + ncol) * HH;
  for (int kb = 0; kb < HH; kb += 32) {
    if (kb + 32 < HH) __builtin_prefetch(xrow + kb + 32, 0, 3);
    // A fragment: row M = m0+ncol, K = kb + {hoffA..hoffA+7, 16+hoffA..}
    v8h alo = *(const v8h*)(xrow + kb + hoffA);
    v8h ahi = *(const v8h*)(xrow + kb + 16 + hoffA);
    v16h afrag = pack16(alo, ahi);
#pragma unroll
    for (int t = 0; t < 4; ++t) {
      // B fragment: col N = n0+t*16+ncol (= W row), K = kb + {hoffA..hoffA+7, 16+hoffA..}
      const _Float16* wrow = Wp + (size_t)(n0 + t * 16 + ncol) * HH + kb + hoffA;
      v8h blo = *(const v8h*)(wrow);
      v8h bhi = *(const v8h*)(wrow + 16);
      acc[t] = wmma_f16(afrag, pack16(blo, bhi), acc[t]);
    }
  }

  // Epilogue: C/D layout -> lane (half, ncol) holds rows r+8*half, col ncol.
#pragma unroll
  for (int t = 0; t < 4; ++t) {
#pragma unroll
    for (int r = 0; r < 8; ++r) {
      int M = m0 + r + 8 * half;          // global row  (b*S + s)
      int N = n0 + t * 16 + ncol;         // global col  (h*64 + d)
      float v = acc[t][r] + bias[N];
      int b = M >> 11, s = M & (SS - 1);
      int h = N >> 6, d = N & (HD - 1);
      size_t bh = (size_t)(b * NH + h);
      _Float16 hv = (_Float16)v;
      if (which == 0)      Q[(bh * SS + s) * HD + d]  = hv;
      else if (which == 1) K[(bh * SS + s) * HD + d]  = hv;
      else                 Vt[(bh * HD + d) * SS + s] = hv;   // transposed
    }
  }
}

// ---------------------------------------------------------------------------
// Fused flash attention.
// Block: 256 threads = 8 waves; wave owns 16 q-rows; block covers 128 q-rows.
// grid = (S/128, NH, B). Loop over k-columns in steps of 32.
// All 8 waves share each 32-column K/V block: the block is async-copied into
// LDS (double-buffered; each wave moves a 512B slice of the next block while
// computing on the current one), cutting L2 traffic 8x.
// Per step: 4 WMMA for S = Q*K^T, online softmax, 4 WMMA for O += P*V.
// P passes through a wave-private LDS tile (D-layout -> A-layout repack).
// ---------------------------------------------------------------------------
__global__ __launch_bounds__(256) void k_flash(
    const _Float16* __restrict__ Q,    // [B*NH][S][HD]
    const _Float16* __restrict__ Km,   // [B*NH][S][HD]
    const _Float16* __restrict__ Vt,   // [B*NH][HD][S]
    const int* __restrict__ amask,     // [B][S]
    float* __restrict__ out) {         // [B][S][H]
  __shared__ _Float16 kbuf[2][32 * HD];   // [buf][kcol][d]   2 x 4KB
  __shared__ _Float16 vbuf[2][HD * 32];   // [buf][d][kcol]   2 x 4KB
  __shared__ _Float16 plds[8][16 * 32];   // per-wave P tile  8KB

  const int lane = threadIdx.x & 31;
  const int wave = threadIdx.x >> 5;
  const int ncol = lane & 15;
  const int half = lane >> 4;

  const int h  = blockIdx.y;
  const int b  = blockIdx.z;
  const int q0 = blockIdx.x * 128 + wave * 16;
  const size_t bh = (size_t)(b * NH + h);

  const _Float16* Qp = Q  + bh * SS * HD;
  const _Float16* Kp = Km + bh * SS * HD;
  const _Float16* Vp = Vt + bh * HD * SS;
  const int* mrow = amask + (size_t)b * SS;

  // -------- async-copy slice assignment (16B per lane per buffer) ---------
  // chunk 0..255 covers one 4KB block linearly.
  const int chunk = wave * 32 + lane;
  const uint32_t k_lds = (uint32_t)(uintptr_t)(&kbuf[0][0]) + chunk * 16;
  const uint32_t v_lds = (uint32_t)(uintptr_t)(&vbuf[0][0]) + chunk * 16;
  // K block at kc is 4KB contiguous (consecutive rows of [s][d]).
  const char* k_g = (const char*)Kp + chunk * 16;
  // V^T rows: d = chunk/4, 16B part = chunk%4, row stride S halves.
  const char* v_g = (const char*)Vp + (size_t)(chunk >> 2) * (SS * 2) +
                    (chunk & 3) * 16;
  const uint32_t bufstride = 32 * HD * 2;  // 4KB

  // -------- Q fragments (A layout), registers for the whole k loop --------
  const _Float16* qrow = Qp + (size_t)(q0 + ncol) * HD;
  v16h qfrag[2];
#pragma unroll
  for (int ds = 0; ds < 2; ++ds) {
    v8h lo = *(const v8h*)(qrow + ds * 32 + half * 8);
    v8h hi = *(const v8h*)(qrow + ds * 32 + 16 + half * 8);
    qfrag[ds] = pack16(lo, hi);
  }

  float m_i[8], l_i[8];
#pragma unroll
  for (int r = 0; r < 8; ++r) { m_i[r] = -1e30f; l_i[r] = 0.0f; }
  v8f o[4] = {};

  const float sc = 0.125f;  // 1/sqrt(64)
  const int NBLK = SS / 32; // 64

  // prologue: stage block 0 into buffer 0
  async_ld_b128(k_lds, k_g);
  async_ld_b128(v_lds, v_g + 0);
  wait_async0();
  __syncthreads();

  for (int i = 0; i < NBLK; ++i) {
    const int kc = i * 32;
    const int bi = i & 1;
    // issue async copy of the NEXT block into the other buffer; it overlaps
    // the compute below and is waited at the bottom of this iteration.
    if (i + 1 < NBLK) {
      async_ld_b128(k_lds + (bi ^ 1) * bufstride,
                    k_g + (size_t)(kc + 32) * (HD * 2));
      async_ld_b128(v_lds + (bi ^ 1) * bufstride, v_g + (kc + 32) * 2);
    }
    const _Float16* kb = &kbuf[bi][0];
    const _Float16* vb = &vbuf[bi][0];

    // ---- scores: two 16-column tiles (B-fragments from LDS) -------------
    v8f sacc[2];
#pragma unroll
    for (int ct = 0; ct < 2; ++ct) {
      v8f s = {};
      const _Float16* krow = kb + (ct * 16 + ncol) * HD;
#pragma unroll
      for (int ds = 0; ds < 2; ++ds) {
        v8h lo = *(const v8h*)(krow + ds * 32 + half * 8);
        v8h hi = *(const v8h*)(krow + ds * 32 + 16 + half * 8);
        s = wmma_f16(qfrag[ds], pack16(lo, hi), s);
      }
      float madd = (1.0f - (float)mrow[kc + ct * 16 + ncol]) * (-1e30f);
#pragma unroll
      for (int r = 0; r < 8; ++r) s[r] = s[r] * sc + madd;
      sacc[ct] = s;
    }

    // ---- online softmax -------------------------------------------------
    float bm[8];
#pragma unroll
    for (int r = 0; r < 8; ++r) bm[r] = fmaxf(sacc[0][r], sacc[1][r]);
#pragma unroll
    for (int r = 0; r < 8; ++r) {
      // row max across the 16 lanes of this half (xor masks stay in-half)
      bm[r] = fmaxf(bm[r], __shfl_xor(bm[r], 1, 32));
      bm[r] = fmaxf(bm[r], __shfl_xor(bm[r], 2, 32));
      bm[r] = fmaxf(bm[r], __shfl_xor(bm[r], 4, 32));
      bm[r] = fmaxf(bm[r], __shfl_xor(bm[r], 8, 32));
    }
    float alpha[8], rsum[8];
#pragma unroll
    for (int r = 0; r < 8; ++r) {
      float mn = fmaxf(m_i[r], bm[r]);
      alpha[r] = __expf(m_i[r] - mn);
      m_i[r] = mn;
      rsum[r] = 0.0f;
    }
    // P = exp(S - m); stage into LDS in D-layout positions
#pragma unroll
    for (int ct = 0; ct < 2; ++ct) {
#pragma unroll
      for (int r = 0; r < 8; ++r) {
        float p = __expf(sacc[ct][r] - m_i[r]);
        rsum[r] += p;
        plds[wave][(r + 8 * half) * 32 + ct * 16 + ncol] = (_Float16)p;
      }
    }
#pragma unroll
    for (int r = 0; r < 8; ++r) {
      rsum[r] += __shfl_xor(rsum[r], 1, 32);
      rsum[r] += __shfl_xor(rsum[r], 2, 32);
      rsum[r] += __shfl_xor(rsum[r], 4, 32);
      rsum[r] += __shfl_xor(rsum[r], 8, 32);
      l_i[r] = l_i[r] * alpha[r] + rsum[r];
    }
#pragma unroll
    for (int t = 0; t < 4; ++t)
#pragma unroll
      for (int r = 0; r < 8; ++r) o[t][r] *= alpha[r];

    // wave-private LDS tile: only this wave reads it back -> counter wait only
    asm volatile("s_wait_dscnt 0" ::: "memory");

    // P as A-fragment (16x32): row M = ncol, K = half*8 + e (+16 for e>=8)
    v8h plo = *(const v8h*)(&plds[wave][ncol * 32 + half * 8]);
    v8h phi = *(const v8h*)(&plds[wave][ncol * 32 + 16 + half * 8]);
    v16h pfrag = pack16(plo, phi);

    // ---- O += P * V  (V^T in LDS: contiguous along k-cols) --------------
#pragma unroll
    for (int t = 0; t < 4; ++t) {
      const _Float16* vrow = vb + (t * 16 + ncol) * 32;
      v8h lo = *(const v8h*)(vrow + half * 8);
      v8h hi = *(const v8h*)(vrow + 16 + half * 8);
      o[t] = wmma_f16(pfrag, pack16(lo, hi), o[t]);
    }

    // drain this iteration's async copies, then publish buffers to all waves
    wait_async0();
    __syncthreads();
  }

  // ---- epilogue: out[b][s][h*64+d] = O / l -------------------------------
#pragma unroll
  for (int t = 0; t < 4; ++t) {
#pragma unroll
    for (int r = 0; r < 8; ++r) {
      int s = q0 + r + 8 * half;
      int d = t * 16 + ncol;
      out[((size_t)b * SS + s) * HH + h * HD + d] = o[t][r] / l_i[r];
    }
  }
}

// ---------------------------------------------------------------------------
// Host-side launcher
// ---------------------------------------------------------------------------
extern "C" void kernel_launch(void* const* d_in, const int* in_sizes, int n_in,
                              void* d_out, int out_size, void* d_ws,
                              size_t ws_size, hipStream_t stream) {
  (void)in_sizes; (void)n_in; (void)out_size; (void)ws_size;

  const float* X  = (const float*)d_in[0];
  const int*   am = (const int*)d_in[1];
  const float* Wq = (const float*)d_in[2];
  const float* bq = (const float*)d_in[3];
  const float* Wk = (const float*)d_in[4];
  const float* bk = (const float*)d_in[5];
  const float* Wv = (const float*)d_in[6];
  const float* bv = (const float*)d_in[7];
  float* out = (float*)d_out;

  // Workspace layout (fp16): X16 | W16[3] | Q16 | K16 | V16t   (~40 MB)
  _Float16* X16  = (_Float16*)d_ws;
  _Float16* W16  = X16 + (size_t)BB * SS * HH;          // 4096*1024
  _Float16* Q16  = W16 + (size_t)3 * HH * HH;
  _Float16* K16  = Q16 + (size_t)BB * SS * HH;
  _Float16* V16t = K16 + (size_t)BB * SS * HH;

  const int nX = BB * SS * HH;   // 4,194,304
  const int nW = HH * HH;        // 1,048,576
  k_cvt<<<4096, 256, 0, stream>>>(X, X16, nX);
  k_cvt<<<1024, 256, 0, stream>>>(Wq, W16, nW);
  k_cvt<<<1024, 256, 0, stream>>>(Wk, W16 + (size_t)nW, nW);
  k_cvt<<<1024, 256, 0, stream>>>(Wv, W16 + (size_t)2 * nW, nW);

  dim3 gproj(BB * SS / 128, HH / 64, 3);
  k_qkv_proj<<<gproj, 256, 0, stream>>>(X16, W16, bq, bk, bv, Q16, K16, V16t);

  dim3 gattn(SS / 128, NH, BB);
  k_flash<<<gattn, 256, 0, stream>>>(Q16, K16, V16t, am, out);
}
